// VQVAE_18588618457488
// MI455X (gfx1250) — matrix operand order repaired and checked by reading, not compiled, not run on hardware
//
#include <hip/hip_runtime.h>

typedef __attribute__((ext_vector_type(16))) __bf16 v16bf;
typedef __attribute__((ext_vector_type(8)))  __bf16 v8bf;
typedef __attribute__((ext_vector_type(8)))  float  v8f;
typedef __attribute__((ext_vector_type(4)))  unsigned uint4v;
typedef __attribute__((ext_vector_type(4)))  int    int4v;
typedef __attribute__((ext_vector_type(8)))  int    int8v;

__device__ __forceinline__ __bf16 f2bf(float f) {
  unsigned u = __builtin_bit_cast(unsigned, f);
  u += 0x7fffu + ((u >> 16) & 1u);               // round-to-nearest-even
  unsigned short h = (unsigned short)(u >> 16);
  return __builtin_bit_cast(__bf16, h);
}

// fast unsigned division by runtime constant: magic = ceil(2^43/d), x < 2^18
__device__ __forceinline__ unsigned fdiv(unsigned x, unsigned long long magic) {
  return (unsigned)(((unsigned long long)x * magic) >> 43);
}

// ---------------------------------------------------------------------------
// TDM: DMA one contiguous 8KB block (2048 x 4B) global -> LDS.
// D# group0: count=1 | lds_addr | global_addr[56:0] | type=2
// D# group1: data_size=4B, tensor_dim0=tile_dim0=2048, tile_dim1=1
// ---------------------------------------------------------------------------
__device__ __forceinline__ void tdm_copy_8k(const void* gsrc, void* ldsdst) {
  unsigned long long ga = (unsigned long long)(uintptr_t)gsrc;
  unsigned ldsoff = (unsigned)(uintptr_t)ldsdst;   // low 32 bits = LDS offset
  uint4v g0 = { 1u,                                 // count = 1 (valid user D#)
                ldsoff,                             // lds_addr
                (unsigned)ga,                       // global_addr[31:0]
                (unsigned)((ga >> 32) & 0x1FFFFFFu) | (2u << 30) }; // [56:32]|type=2
  int8v g1 = { (int)(2u << 16),      // workgroup_mask=0, data_size=2 (4 bytes)
               (int)0x08000000u,     // tensor_dim0[15:0]=2048 in word1[31:16]
               (int)(1u << 16),      // tensor_dim0 hi=0 ; tensor_dim1 lo=1
               (int)0x08000000u,     // tensor_dim1 hi=0 ; tile_dim0=2048
               1,                    // tile_dim1 = 1
               2048,                 // tensor_dim0_stride = 2048 elements
               0, 0 };
  int4v z4 = { 0, 0, 0, 0 };
#if defined(__HIP_DEVICE_COMPILE__)
#if __clang_major__ >= 23
  int8v z8 = { 0, 0, 0, 0, 0, 0, 0, 0 };
  __builtin_amdgcn_tensor_load_to_lds(g0, g1, z4, z4, z8, 0);
#else
  __builtin_amdgcn_tensor_load_to_lds(g0, g1, z4, z4, 0);
#endif
  __builtin_amdgcn_s_wait_tensorcnt(0);
#endif
  (void)g0; (void)g1; (void)z4;
}

// ---------------------------------------------------------------------------
// WMMA GEMM: C[M,Cout] = A[M,K] * B[K,N] (+bias)(+residual)
//   MODE 0: A is plain fp32 matrix (lda), batched via blockIdx.z strides
//   MODE 1: A is implicit im2col of NHWC fp32 tensor (conv KSZ, stride/pad)
//   MODE 2: A is implicit im2col of 4x4/s2/p1 transposed-conv gather
// B is bf16, pre-swizzled into TDM/LDS tile layout:
//   block = [ (n0/64)*(Kpad/32) + (k0/32) ] * 2048 + n_in_tile*32 + k_in_tile
// so K-tiles for one N-block are contiguous -> one 8KB TDM copy per K-step(64).
// ---------------------------------------------------------------------------
struct GemmP {
  const float*  A;
  const __bf16* Bm;
  const float*  bias;
  const float*  radd;
  float*        C;
  int M, K, Kpad, Npad, Cout, lda;
  int H, W, Cin, Ho, Wo, stride, pad;
  int zshift;
  long long sA1, sA2, sB, sC;
  unsigned long long mHoWo, mWo;    // magic reciprocals (shift 43)
};

template <int MODE, int KSZ>
__global__ __launch_bounds__(256) void wmma_gemm_kernel(GemmP p) {
  __shared__ __align__(64) __bf16 As[2][64][32];   // two K-panels, [m][k]
  __shared__ __align__(64) __bf16 Bs[2][64][32];   // two K-panels, [n][k]
  const int m0 = blockIdx.x * 64;
  const int n0 = blockIdx.y * 64;
  const int z  = blockIdx.z;
  const long long offA = (long long)(z >> p.zshift) * p.sA1 +
                         (long long)(z & ((1 << p.zshift) - 1)) * p.sA2;
  const __bf16* Bmz = p.Bm + (long long)z * p.sB;
  float*        Cz  = p.C  + (long long)z * p.sC;
  const float*  raddz = p.radd ? (p.radd + (long long)z * p.sC) : nullptr;

  const int tid  = threadIdx.x;
  const int lane = tid & 31;          // wave32
  const int wave = tid >> 5;          // 8 waves
  const int mi   = wave & 3;          // M-subtile 0..3
  const int nb0  = (wave >> 2) * 2;   // first of two N-subtiles
  const int lrow = lane & 15;
  const int lhi  = lane >> 4;
  const int HoWo = p.Ho * p.Wo;

  // Each thread stages one A row (mm = tid/4), 16 consecutive k values.
  const int mm    = tid >> 2;
  const int kbase = (tid & 3) * 16;
  const int m     = m0 + mm;
  int ab = 0, aho = 0, awo = 0;
  if (MODE != 0) {
    ab  = fdiv((unsigned)m, p.mHoWo);
    int rem = m - ab * HoWo;
    aho = fdiv((unsigned)rem, p.mWo);
    awo = rem - aho * p.Wo;
  }
  const __bf16* Bpanel = Bmz + ((long long)blockIdx.y * (p.Kpad >> 5)) * 2048;

  v8f acc0 = {};
  v8f acc1 = {};

  const int ksteps = p.Kpad >> 6;            // 64 K per iteration
  for (int kt = 0; kt < ksteps; ++kt) {
    const int k0 = kt << 6;
    __syncthreads();
    // ---- stage A (64x64), implicit im2col gather, fp32 -> bf16 ----
#pragma unroll
    for (int i = 0; i < 16; ++i) {
      int kk = kbase + i, kg = k0 + kk;
      float v = 0.0f;
      if (m < p.M && kg < p.K) {
        if (MODE == 0) {
          v = p.A[offA + (long long)m * p.lda + kg];
        } else {
          int ci = kg / (KSZ * KSZ);
          int r  = kg - ci * (KSZ * KSZ);
          int fy = r / KSZ, fx = r - fy * KSZ;
          int iy, ix; bool ok;
          if (MODE == 1) {
            iy = aho * p.stride - p.pad + fy;
            ix = awo * p.stride - p.pad + fx;
            ok = (iy >= 0) & (iy < p.H) & (ix >= 0) & (ix < p.W);
          } else {                             // transposed conv (s=2,p=1,k=4)
            int ny = aho + p.pad - fy, nx = awo + p.pad - fx;
            iy = ny >> 1; ix = nx >> 1;
            ok = (ny >= 0) & (nx >= 0) & ((ny & 1) == 0) & ((nx & 1) == 0) &
                 (iy < p.H) & (ix < p.W);
          }
          if (ok) v = p.A[(((long long)ab * p.H + iy) * p.W + ix) * p.Cin + ci];
        }
      }
      As[kk >> 5][mm][kk & 31] = f2bf(v);
    }
    // ---- stage B: one 8KB contiguous block via Tensor Data Mover ----
    if (wave == 0) {
      tdm_copy_8k(Bpanel + (long long)(kt * 2) * 2048, &Bs[0][0][0]);
    }
    __syncthreads();

    // ---- 2 K-panels x 2 N-subtiles = 4 WMMAs per barrier round ----
#pragma unroll
    for (int pnl = 0; pnl < 2; ++pnl) {
      const int arow = mi * 16 + lrow;
      v16bf afrag = *(const v16bf*)&As[pnl][arow][16 * lhi];
      v16bf bf0 = *(const v16bf*)&Bs[pnl][nb0 * 16 + lrow][16 * lhi];
      v16bf bf1 = *(const v16bf*)&Bs[pnl][(nb0 + 1) * 16 + lrow][16 * lhi];
#if defined(__HIP_DEVICE_COMPILE__)
      acc0 = __builtin_amdgcn_wmma_f32_16x16x32_bf16(false, afrag, false, bf0,
                                                     (short)0, acc0, false, false);
      acc1 = __builtin_amdgcn_wmma_f32_16x16x32_bf16(false, afrag, false, bf1,
                                                     (short)0, acc1, false, false);
#endif
    }
  }

  // ---- epilogue: bias + optional residual, NHWC store (ldc == Cout) ----
#pragma unroll
  for (int t = 0; t < 2; ++t) {
    int n = n0 + (nb0 + t) * 16 + lrow;
    if (n >= p.Cout) continue;
    float bsv = p.bias ? p.bias[n] : 0.0f;
#pragma unroll
    for (int r = 0; r < 8; ++r) {
      int mo = m0 + mi * 16 + lhi * 8 + r;
      if (mo < p.M) {
        float v = (t ? acc1[r] : acc0[r]) + bsv;
        long long idx = (long long)mo * p.Cout + n;
        if (raddz) v += raddz[idx];
        Cz[idx] = v;
      }
    }
  }
}

// ---------------------------------------------------------------------------
// Weight packing (fp32 -> bf16, TDM tile layout), repacked every call
// ---------------------------------------------------------------------------
__device__ __forceinline__ long long tile_idx(int k, int n, int Kpad) {
  return ((long long)(n >> 6) * (Kpad >> 5) + (k >> 5)) * 2048 +
         (long long)(n & 63) * 32 + (k & 31);
}

__global__ __launch_bounds__(256) void pack_conv_w_kernel(const float* w, __bf16* dst,
    int K, int Kpad, int Cout, int Npad, int Cin, int ksz, int deconv) {
  int idx = blockIdx.x * 256 + threadIdx.x;
  if (idx >= Kpad * Npad) return;
  int k = idx / Npad, n = idx - k * Npad;
  float v = 0.0f;
  if (k < K && n < Cout) {
    int kk2 = ksz * ksz;
    int ci = k / kk2, r = k - ci * kk2;
    int fy = r / ksz, fx = r - fy * ksz;
    long long wi = deconv ? ((((long long)ci * Cout + n) * ksz + fy) * ksz + fx)
                          : ((((long long)n * Cin + ci) * ksz + fy) * ksz + fx);
    v = w[wi];
  }
  dst[tile_idx(k, n, Kpad)] = f2bf(v);
}

__global__ __launch_bounds__(256) void pack_linear_w_kernel(const float* w, __bf16* dst,
    int K, int Kpad, int N, int Npad) {
  int idx = blockIdx.x * 256 + threadIdx.x;
  if (idx >= Kpad * Npad) return;
  int k = idx / Npad, n = idx - k * Npad;
  float v = (k < K && n < N) ? w[(long long)n * K + k] : 0.0f;
  dst[tile_idx(k, n, Kpad)] = f2bf(v);
}

// k^T and v packed per (batch,head) as tiled bf16 B-matrices
//  kT: [K=64 (d)][N=256 (s)]   vB: [K=256 (s)][N=64 (d)]
__global__ __launch_bounds__(256) void pack_kT_v_kernel(const float* qkv,
                                                        __bf16* kT, __bf16* vB) {
  int idx = blockIdx.x * 256 + threadIdx.x;
  if (idx >= 64 * 64 * 256) return;
  int s  = idx % 256;
  int d  = (idx / 256) % 64;
  int bh = idx / (256 * 64);
  int b = bh >> 2, h = bh & 3;
  long long src = ((long long)(b * 256 + s)) * 768 + h * 64 + d;
  kT[(long long)bh * 64 * 256 + tile_idx(d, s, 64)]  = f2bf(qkv[src + 256]);
  vB[(long long)bh * 256 * 64 + tile_idx(s, d, 256)] = f2bf(qkv[src + 512]);
}

__global__ __launch_bounds__(256) void merge_heads_kernel(const float* att, float* out) {
  int idx = blockIdx.x * 256 + threadIdx.x;
  if (idx >= 16 * 256 * 256) return;
  int e = idx % 256;
  int s = (idx / 256) % 256;
  int b = idx / (256 * 256);
  int h = e / 64, d = e - h * 64;
  out[idx] = att[(((long long)(b * 4 + h) * 256) + s) * 64 + d];
}

// ---------------------------------------------------------------------------
// GroupNorm (+optional SiLU) over NHWC, one block per (batch, group)
// ---------------------------------------------------------------------------
__global__ __launch_bounds__(256) void groupnorm_kernel(const float* x, float* y,
    const float* gamma, const float* beta, int HW, int C, int groups, int do_silu) {
  int g = blockIdx.x % groups;
  int b = blockIdx.x / groups;
  int Cg = C / groups;
  long long base = (long long)b * HW * C + (long long)g * Cg;
  int n = HW * Cg;
  __shared__ float ssum[256], ssq[256];
  float s = 0.f, sq = 0.f;
  for (int i = threadIdx.x; i < n; i += 256) {
    int hw = i / Cg, c = i - hw * Cg;
    float v = x[base + (long long)hw * C + c];
    s += v; sq += v * v;
  }
  ssum[threadIdx.x] = s; ssq[threadIdx.x] = sq;
  __syncthreads();
  for (int off = 128; off > 0; off >>= 1) {
    if (threadIdx.x < off) {
      ssum[threadIdx.x] += ssum[threadIdx.x + off];
      ssq[threadIdx.x]  += ssq[threadIdx.x + off];
    }
    __syncthreads();
  }
  float mean = ssum[0] / n;
  float var  = ssq[0] / n - mean * mean;
  float rstd = rsqrtf(var + 1e-5f);
  for (int i = threadIdx.x; i < n; i += 256) {
    int hw = i / Cg, c = i - hw * Cg;
    long long idx = base + (long long)hw * C + c;
    int ch = g * Cg + c;
    float v = (x[idx] - mean) * rstd * gamma[ch] + beta[ch];
    if (do_silu) v = v / (1.0f + __expf(-v));
    y[idx] = v;
  }
}

__global__ __launch_bounds__(256) void softmax256_kernel(float* a, float scale) {
  float* p = a + (long long)blockIdx.x * 256;
  int t = threadIdx.x;
  float v = p[t] * scale;
  __shared__ float sh[256];
  sh[t] = v; __syncthreads();
  for (int off = 128; off > 0; off >>= 1) {
    if (t < off) sh[t] = fmaxf(sh[t], sh[t + off]);
    __syncthreads();
  }
  float mx = sh[0];
  __syncthreads();
  float e = __expf(v - mx);
  sh[t] = e; __syncthreads();
  for (int off = 128; off > 0; off >>= 1) {
    if (t < off) sh[t] += sh[t + off];
    __syncthreads();
  }
  p[t] = e / sh[0];
}

// ---------------------------------------------------------------------------
// Vector quantizer: argmin over 8192 codes (dim 4) + fused loss reduction
// ---------------------------------------------------------------------------
__global__ __launch_bounds__(256) void quantize_kernel(const float* z, const float* cb,
    float* q_nhwc, float* q_nchw, float* acc, int M, int ncodes) {
  int m = blockIdx.x * 256 + threadIdx.x;
  float sq = 0.0f;
  if (m < M) {
    float z0 = z[m*4+0], z1 = z[m*4+1], z2 = z[m*4+2], z3 = z[m*4+3];
    int best = 0; float bd = 3.4e38f;
    for (int c = 0; c < ncodes; ++c) {
      float d0 = z0 - cb[c*4+0], d1 = z1 - cb[c*4+1];
      float d2 = z2 - cb[c*4+2], d3 = z3 - cb[c*4+3];
      float d = d0*d0 + d1*d1 + d2*d2 + d3*d3;
      if (d < bd) { bd = d; best = c; }
    }
    int b = m >> 8, hw = m & 255;
    for (int c = 0; c < 4; ++c) {
      float q = cb[best*4+c];
      q_nhwc[m*4+c] = q;
      q_nchw[((long long)b*4 + c)*256 + hw] = q;
      float dd = q - z[m*4+c];
      sq += dd * dd;
    }
  }
  __shared__ float sh[256];
  sh[threadIdx.x] = sq;
  __syncthreads();
  for (int off = 128; off > 0; off >>= 1) {
    if (threadIdx.x < off) sh[threadIdx.x] += sh[threadIdx.x + off];
    __syncthreads();
  }
  if (threadIdx.x == 0) atomicAdd(acc, sh[0]);
}

__global__ void zero_kernel(float* p, int n) {
  int i = blockIdx.x * blockDim.x + threadIdx.x;
  if (i < n) p[i] = 0.0f;
}
__global__ void finalize_losses_kernel(const float* acc, float* out, float inv) {
  if (threadIdx.x == 0) { float l = acc[0] * inv; out[0] = l; out[1] = l; }
}

__global__ __launch_bounds__(256) void nchw_to_nhwc_kernel(const float* src, float* dst,
                                                           int C, int H, int W) {
  long long idx = (long long)blockIdx.x * 256 + threadIdx.x;
  long long total = 16ll * C * H * W;
  if (idx >= total) return;
  long long r = idx;
  int c = (int)(r % C); r /= C;
  int w = (int)(r % W); r /= W;
  int h = (int)(r % H); r /= H;
  int b = (int)r;
  dst[idx] = src[(((long long)b * C + c) * H + h) * W + w];
}
__global__ __launch_bounds__(256) void nhwc_to_nchw_kernel(const float* src, float* dst,
                                                           int C, int H, int W) {
  long long idx = (long long)blockIdx.x * 256 + threadIdx.x;
  long long total = 16ll * C * H * W;
  if (idx >= total) return;
  long long r = idx;
  int w = (int)(r % W); r /= W;
  int h = (int)(r % H); r /= H;
  int c = (int)(r % C); r /= C;
  int b = (int)r;
  dst[idx] = src[(((long long)b * H + h) * W + w) * C + c];
}

// ---------------------------------------------------------------------------
// Host orchestration
// ---------------------------------------------------------------------------
extern "C" void kernel_launch(void* const* d_in, const int* in_sizes, int n_in,
                              void* d_out_, int out_size, void* d_ws, size_t ws_size,
                              hipStream_t stream) {
  (void)in_sizes; (void)n_in; (void)out_size; (void)ws_size;
  float* d_out = (float*)d_out_;

  char* wsc = (char*)d_ws;
  const size_t BUFB = 16ull * 128 * 128 * 128 * sizeof(float);   // 128 MB
  float*  bufA  = (float*)wsc;  wsc += BUFB;
  float*  bufB  = (float*)wsc;  wsc += BUFB;
  float*  bufC  = (float*)wsc;  wsc += BUFB;
  __bf16* wpack = (__bf16*)wsc; wsc += (8ull << 20);
  __bf16* kTb   = (__bf16*)wsc; wsc += 64ull * 64 * 256 * sizeof(__bf16);
  __bf16* vb    = (__bf16*)wsc; wsc += 64ull * 256 * 64 * sizeof(__bf16);
  float*  lossacc = (float*)wsc; wsc += 256;

  // ---- walk params in setup_inputs() flatten order ----
  int pi = 0;
  auto P = [&]() { return (const float*)d_in[pi++]; };
  const float* x = P();
  const float* enc_in_w = P(); const float* enc_in_b = P();
  struct ResP  { const float *n1g,*n1b,*c1w,*c1b,*n2g,*n2b,*c2w,*c2b,*rw,*rb; };
  struct AttnP { const float *ng,*nb,*wqkv,*bqkv,*wo,*bo; };
  auto readRes = [&]() {
    ResP r; r.n1g=P(); r.n1b=P(); r.c1w=P(); r.c1b=P(); r.n2g=P(); r.n2b=P();
    r.c2w=P(); r.c2b=P(); r.rw=P(); r.rb=P(); return r;
  };
  auto readAttn = [&]() {
    AttnP a; a.ng=P(); a.nb=P(); a.wqkv=P(); a.bqkv=P(); a.wo=P(); a.bo=P(); return a;
  };
  ResP edres[3][2]; const float* eddw[3]; const float* eddb[3];
  for (int i = 0; i < 3; ++i) {
    edres[i][0] = readRes(); edres[i][1] = readRes(); eddw[i] = P(); eddb[i] = P();
  }
  ResP emres[3]; AttnP emattn[2];
  for (int i = 0; i < 3; ++i) emres[i] = readRes();
  for (int i = 0; i < 2; ++i) emattn[i] = readAttn();
  const float* enc_ng = P();    const float* enc_nb = P();
  const float* enc_out_w = P(); const float* enc_out_b = P();
  const float* pre_q_w = P();   const float* pre_q_b = P();
  const float* codebook = P();
  const float* post_q_w = P();  const float* post_q_b = P();
  const float* dec_in_w = P();  const float* dec_in_b = P();
  ResP dmres[3]; AttnP dmattn[2];
  for (int i = 0; i < 3; ++i) dmres[i] = readRes();
  for (int i = 0; i < 2; ++i) dmattn[i] = readAttn();
  const float* duuw[3]; const float* duub[3]; ResP dures[3][2];
  for (int i = 0; i < 3; ++i) {
    duuw[i] = P(); duub[i] = P(); dures[i][0] = readRes(); dures[i][1] = readRes();
  }
  const float* dec_ng = P();    const float* dec_nb = P();
  const float* dec_out_w = P(); const float* dec_out_b = P();

  // ---- op helpers ----
  float* cur = bufA; float* t1 = bufB; float* t2 = bufC;
  auto swp = [](float*& a, float*& b) { float* t = a; a = b; b = t; };
  auto mkmagic = [](unsigned d) -> unsigned long long {
    if (d == 0) d = 1;
    return (unsigned long long)((((__uint128_t)1 << 43) + d - 1) / d);
  };

  auto run_gemm = [&](GemmP& p, int mode, int ksz, int zb) {
    dim3 g((unsigned)((p.M + 63) / 64), (unsigned)(p.Npad / 64), (unsigned)zb);
    dim3 blk(256);
    if (mode == 0)        wmma_gemm_kernel<0, 1><<<g, blk, 0, stream>>>(p);
    else if (mode == 1) {
      if (ksz == 1)       wmma_gemm_kernel<1, 1><<<g, blk, 0, stream>>>(p);
      else if (ksz == 3)  wmma_gemm_kernel<1, 3><<<g, blk, 0, stream>>>(p);
      else                wmma_gemm_kernel<1, 4><<<g, blk, 0, stream>>>(p);
    } else                wmma_gemm_kernel<2, 4><<<g, blk, 0, stream>>>(p);
  };

  auto conv = [&](const float* in, float* out, const float* w, const float* bias,
                  const float* radd, int H, int Wd, int Ci, int Co,
                  int k, int s, int pd, int mode) {
    int Ho = (mode == 2) ? H * 2  : (H  + 2 * pd - k) / s + 1;
    int Wo = (mode == 2) ? Wd * 2 : (Wd + 2 * pd - k) / s + 1;
    int K = Ci * k * k;
    int Kpad = (K + 63) & ~63;
    int Npad = (Co + 63) & ~63;
    int tot = Kpad * Npad;
    pack_conv_w_kernel<<<dim3((tot + 255) / 256), dim3(256), 0, stream>>>(
        w, wpack, K, Kpad, Co, Npad, Ci, k, (mode == 2) ? 1 : 0);
    GemmP p = {};
    p.A = in; p.Bm = wpack; p.bias = bias; p.radd = radd; p.C = out;
    p.M = 16 * Ho * Wo; p.K = K; p.Kpad = Kpad; p.Npad = Npad; p.Cout = Co;
    p.lda = 0;
    p.H = H; p.W = Wd; p.Cin = Ci; p.Ho = Ho; p.Wo = Wo; p.stride = s; p.pad = pd;
    p.zshift = 0; p.sA1 = 0; p.sA2 = 0; p.sB = 0; p.sC = 0;
    p.mHoWo = mkmagic((unsigned)(Ho * Wo)); p.mWo = mkmagic((unsigned)Wo);
    run_gemm(p, mode, k, 1);
  };

  auto gn = [&](const float* in, float* out, const float* g, const float* b,
                int HW, int C, int silu) {
    groupnorm_kernel<<<dim3(16 * 32), dim3(256), 0, stream>>>(in, out, g, b, HW, C, 32, silu);
  };

  auto resnet = [&](const ResP& r, int Ci, int Co, int H) {
    gn(cur, t2, r.n1g, r.n1b, H * H, Ci, 1);                      // t2 = silu(gn1(x))
    conv(cur, t1, r.rw, r.rb, nullptr, H, H, Ci, Co, 1, 1, 0, 1); // t1 = skip(x)
    conv(t2, cur, r.c1w, r.c1b, nullptr, H, H, Ci, Co, 3, 1, 1, 1);
    gn(cur, t2, r.n2g, r.n2b, H * H, Co, 1);
    conv(t2, cur, r.c2w, r.c2b, t1, H, H, Co, Co, 3, 1, 1, 1);    // + skip (fused)
  };

  auto attn = [&](const AttnP& a) {
    gn(cur, t1, a.ng, a.nb, 256, 256, 0);
    // qkv = h @ wqkv^T + bqkv      (M=4096, K=256, N=768)
    pack_linear_w_kernel<<<dim3((256 * 768 + 255) / 256), dim3(256), 0, stream>>>(
        a.wqkv, wpack, 256, 256, 768, 768);
    GemmP pq = {};
    pq.A = t1; pq.Bm = wpack; pq.bias = a.bqkv; pq.C = t2;
    pq.M = 16 * 256; pq.K = 256; pq.Kpad = 256; pq.Npad = 768; pq.Cout = 768;
    pq.lda = 256; pq.Ho = 1; pq.Wo = 1;
    pq.zshift = 0; pq.sA1 = 0; pq.sA2 = 0; pq.sB = 0; pq.sC = 0;
    pq.mHoWo = mkmagic(1); pq.mWo = mkmagic(1);
    run_gemm(pq, 0, 1, 1);
    pack_kT_v_kernel<<<dim3((64 * 64 * 256 + 255) / 256), dim3(256), 0, stream>>>(t2, kTb, vb);
    // scores = q @ k^T per (b,h)   (64 batches, M=256, K=64, N=256)
    GemmP ps = {};
    ps.A = t2; ps.Bm = kTb; ps.C = t1;
    ps.M = 256; ps.K = 64; ps.Kpad = 64; ps.Npad = 256; ps.Cout = 256;
    ps.lda = 768; ps.Ho = 1; ps.Wo = 1;
    ps.zshift = 2; ps.sA1 = 256ll * 768; ps.sA2 = 64; ps.sB = 64ll * 256; ps.sC = 256ll * 256;
    ps.mHoWo = mkmagic(1); ps.mWo = mkmagic(1);
    run_gemm(ps, 0, 1, 64);
    softmax256_kernel<<<dim3(64 * 256), dim3(256), 0, stream>>>(t1, 0.125f);
    // out = softmax(scores) @ v    (64 batches, M=256, K=256, N=64)
    GemmP pv = {};
    pv.A = t1; pv.Bm = vb; pv.C = t2;
    pv.M = 256; pv.K = 256; pv.Kpad = 256; pv.Npad = 64; pv.Cout = 64;
    pv.lda = 256; pv.Ho = 1; pv.Wo = 1;
    pv.zshift = 0; pv.sA1 = 256ll * 256; pv.sA2 = 0; pv.sB = 256ll * 64; pv.sC = 256ll * 64;
    pv.mHoWo = mkmagic(1); pv.mWo = mkmagic(1);
    run_gemm(pv, 0, 1, 64);
    merge_heads_kernel<<<dim3((16 * 256 * 256 + 255) / 256), dim3(256), 0, stream>>>(t2, t1);
    // proj + residual (fused radd)
    pack_linear_w_kernel<<<dim3((256 * 256 + 255) / 256), dim3(256), 0, stream>>>(
        a.wo, wpack, 256, 256, 256, 256);
    GemmP pp = {};
    pp.A = t1; pp.Bm = wpack; pp.bias = a.bo; pp.radd = cur; pp.C = cur;
    pp.M = 16 * 256; pp.K = 256; pp.Kpad = 256; pp.Npad = 256; pp.Cout = 256;
    pp.lda = 256; pp.Ho = 1; pp.Wo = 1;
    pp.zshift = 0; pp.sA1 = 0; pp.sA2 = 0; pp.sB = 0; pp.sC = 0;
    pp.mHoWo = mkmagic(1); pp.mWo = mkmagic(1);
    run_gemm(pp, 0, 1, 1);
  };

  auto mid = [&](const ResP* rs, const AttnP* as) {
    resnet(rs[0], 256, 256, 16);
    for (int i = 0; i < 2; ++i) { attn(as[i]); resnet(rs[i + 1], 256, 256, 16); }
  };

  // ---- pipeline ----
  zero_kernel<<<dim3(1), dim3(64), 0, stream>>>(lossacc, 8);
  nchw_to_nhwc_kernel<<<dim3((16 * 3 * 128 * 128 + 255) / 256), dim3(256), 0, stream>>>(
      x, cur, 3, 128, 128);
  conv(cur, t1, enc_in_w, enc_in_b, nullptr, 128, 128, 3, 64, 3, 1, 1, 1); swp(cur, t1);

  int dch[4] = {64, 128, 256, 256};
  int Hc = 128;
  for (int i = 0; i < 3; ++i) {
    int ci = dch[i], co = dch[i + 1];
    resnet(edres[i][0], ci, co, Hc);
    resnet(edres[i][1], co, co, Hc);
    conv(cur, t1, eddw[i], eddb[i], nullptr, Hc, Hc, co, co, 4, 2, 1, 1); swp(cur, t1);
    Hc /= 2;
  }
  mid(emres, emattn);
  gn(cur, t1, enc_ng, enc_nb, 256, 256, 1); swp(cur, t1);
  conv(cur, t1, enc_out_w, enc_out_b, nullptr, 16, 16, 256, 4, 3, 1, 1, 1); swp(cur, t1);
  conv(cur, t1, pre_q_w, pre_q_b, nullptr, 16, 16, 4, 4, 1, 1, 0, 1); swp(cur, t1);

  quantize_kernel<<<dim3(16), dim3(256), 0, stream>>>(
      cur, codebook, t1, d_out + 786432, lossacc, 4096, 8192);
  swp(cur, t1);

  conv(cur, t1, post_q_w, post_q_b, nullptr, 16, 16, 4, 4, 1, 1, 0, 1); swp(cur, t1);
  conv(cur, t1, dec_in_w, dec_in_b, nullptr, 16, 16, 4, 256, 3, 1, 1, 1); swp(cur, t1);
  mid(dmres, dmattn);

  int uci[3] = {256, 256, 128}, uco[3] = {256, 128, 64};
  Hc = 16;
  for (int i = 0; i < 3; ++i) {
    conv(cur, t1, duuw[i], duub[i], nullptr, Hc, Hc, uci[i], uci[i], 4, 2, 1, 2); swp(cur, t1);
    Hc *= 2;
    resnet(dures[i][0], uci[i], uco[i], Hc);
    resnet(dures[i][1], uco[i], uco[i], Hc);
  }
  gn(cur, t1, dec_ng, dec_nb, 128 * 128, 64, 1); swp(cur, t1);
  conv(cur, t1, dec_out_w, dec_out_b, nullptr, 128, 128, 64, 3, 3, 1, 1, 1); swp(cur, t1);
  nhwc_to_nchw_kernel<<<dim3((16 * 3 * 128 * 128 + 255) / 256), dim3(256), 0, stream>>>(
      cur, d_out, 3, 128, 128);
  finalize_losses_kernel<<<dim3(1), dim3(1), 0, stream>>>(
      lossacc, d_out + 786432 + 16384, 1.0f / 16384.0f);
}
